// CausalSelfAttention_25194278158656
// MI455X (gfx1250) — matrix operand order repaired and checked by reading, not compiled, not run on hardware
//
#include <hip/hip_runtime.h>
#include <hip/hip_bf16.h>

// ---------------------------------------------------------------------------
// CDNA5 (gfx1250) causal self-attention pipeline: bf16 WMMA + TDM staging.
// ---------------------------------------------------------------------------

typedef __attribute__((ext_vector_type(16))) __bf16 v16bf;
typedef __attribute__((ext_vector_type(8)))  float  v8f;

union Frag16 { v16bf v; uint4 q[2]; unsigned short u[16]; };

#if defined(__has_builtin)
#  if __has_builtin(__builtin_amdgcn_tensor_load_to_lds)
#    define USE_TDM 1
#  endif
#endif
#ifndef USE_TDM
#  define USE_TDM 0
#endif

#if USE_TDM
typedef unsigned int u32x4 __attribute__((ext_vector_type(4)));
typedef int          i32x8 __attribute__((ext_vector_type(8)));
typedef int          i32x4 __attribute__((ext_vector_type(4)));

// Issue a TDM load of a 2D bf16 tile (tile_d0 elems/row, tile_d1 rows, row
// stride in elements) from global memory into LDS at byte offset lds_off.
// Tensor dims are set to the tile dims (all tiles are interior; no OOB).
static __device__ __forceinline__ void tdm_load_tile_2d(
    unsigned lds_off, const unsigned short* gptr,
    unsigned tile_d0, unsigned tile_d1, unsigned row_stride_elems) {
  unsigned long long ga = (unsigned long long)(const void*)gptr;
  u32x4 g0;
  g0[0] = 1u;                                            // count=1, user mode
  g0[1] = lds_off;                                       // D#.lds_addr
  g0[2] = (unsigned)ga;                                  // global_addr[31:0]
  g0[3] = (unsigned)((ga >> 32) & 0x01FFFFFFull) | (2u << 30);  // ga[56:32] | type=2
  i32x8 g1;
  g1[0] = (int)(1u << 16);                               // data_size = 2 bytes
  g1[1] = (int)((tile_d0 & 0xFFFFu) << 16);              // tensor_dim0[15:0]
  g1[2] = (int)((tile_d0 >> 16) | ((tile_d1 & 0xFFFFu) << 16)); // td0 hi | td1 lo
  g1[3] = (int)((tile_d1 >> 16) | (tile_d0 << 16));      // td1 hi | tile_dim0
  g1[4] = (int)(tile_d1 & 0xFFFFu);                      // tile_dim1 (tile_dim2=0)
  g1[5] = (int)row_stride_elems;                         // tensor_dim0_stride[31:0]
  g1[6] = 0;
  g1[7] = 0;
  i32x4 z4 = {0, 0, 0, 0};
  i32x8 z8 = {0, 0, 0, 0, 0, 0, 0, 0};
  __builtin_amdgcn_tensor_load_to_lds(g0, g1, z4, z4, z8, 0);
}

static __device__ __forceinline__ unsigned lds_offset_of(const void* p) {
  return (unsigned)(size_t)p;   // flat LDS address: low 32 bits = LDS offset
}
#endif

static __device__ __forceinline__ unsigned short f2bf(float f) {
  unsigned u = __float_as_uint(f);
  u += 0x7FFFu + ((u >> 16) & 1u);        // round-to-nearest-even
  return (unsigned short)(u >> 16);
}
static __device__ __forceinline__ float bf2f(unsigned short h) {
  return __uint_as_float(((unsigned)h) << 16);
}

// ---------------------------------------------------------------------------
// f32 -> bf16 conversion, 4 elements per thread
// ---------------------------------------------------------------------------
__global__ __launch_bounds__(256) void cvt_f32_bf16(const float* __restrict__ src,
                                                    unsigned short* __restrict__ dst,
                                                    int n4) {
  int i = blockIdx.x * 256 + threadIdx.x;
  if (i >= n4) return;
  float4 f = ((const float4*)src)[i];
  unsigned short o[4] = { f2bf(f.x), f2bf(f.y), f2bf(f.z), f2bf(f.w) };
  ((uint2*)dst)[i] = *(const uint2*)o;
}

// ---------------------------------------------------------------------------
// GEMM: C[M,N] = A[M,K] * Bt[N,K]^T   (bf16 in, f32 accum)
// 256 threads (8 waves), tile 128(M) x 64(N), K staged 32-wide.
// TDM path: double-buffered LDS; wave 0 DMAs slab k+1 while all compute k.
// ---------------------------------------------------------------------------
__global__ __launch_bounds__(256) void gemm_bf16_nt(
    const unsigned short* __restrict__ A,
    const unsigned short* __restrict__ Bt,
    float* __restrict__ Cf,
    unsigned short* __restrict__ Cb,
    int M, int N, int K)
{
#if USE_TDM
  __shared__ __align__(16) unsigned short As[2][128 * 32];
  __shared__ __align__(16) unsigned short Bs[2][64 * 32];
#else
  __shared__ __align__(16) unsigned short As[1][128 * 32];
  __shared__ __align__(16) unsigned short Bs[1][64 * 32];
#endif

  int tid  = threadIdx.x;
  int wave = tid >> 5, lane = tid & 31;
  int n16  = lane & 15, half = lane >> 4;
  int m0   = blockIdx.y * 128, n0 = blockIdx.x * 64;

  v8f acc[4] = {};
  int cur = 0;

#if USE_TDM
  if (wave == 0) {
    tdm_load_tile_2d(lds_offset_of(&As[0][0]), A + (size_t)m0 * K, 32, 128, K);
    tdm_load_tile_2d(lds_offset_of(&Bs[0][0]), Bt + (size_t)n0 * K, 32, 64, K);
  }
#endif

  for (int k0 = 0; k0 < K; k0 += 32) {
#if USE_TDM
    __builtin_amdgcn_s_wait_tensorcnt(0);   // this slab's DMA complete (issuer)
    __syncthreads();
    if (wave == 0 && k0 + 32 < K) {        // prefetch next slab into other buf
      tdm_load_tile_2d(lds_offset_of(&As[cur ^ 1][0]),
                       A + (size_t)m0 * K + (k0 + 32), 32, 128, K);
      tdm_load_tile_2d(lds_offset_of(&Bs[cur ^ 1][0]),
                       Bt + (size_t)n0 * K + (k0 + 32), 32, 64, K);
    }
#else
#pragma unroll
    for (int i = 0; i < 4; ++i) {
      int e = tid * 4 + i * 1024;
      int r = e >> 5, c = e & 31;
      *(uint2*)&As[0][e] = *(const uint2*)&A[(size_t)(m0 + r) * K + k0 + c];
    }
#pragma unroll
    for (int i = 0; i < 2; ++i) {
      int e = tid * 4 + i * 1024;
      int r = e >> 5, c = e & 31;
      *(uint2*)&Bs[0][e] = *(const uint2*)&Bt[(size_t)(n0 + r) * K + k0 + c];
    }
    __syncthreads();
#endif

    // A fragment (16x32): lane -> row = n16, 16 contiguous K at half*16
    Frag16 af;
    af.q[0] = *(const uint4*)&As[cur][(wave * 16 + n16) * 32 + half * 16];
    af.q[1] = *(const uint4*)&As[cur][(wave * 16 + n16) * 32 + half * 16 + 8];

#pragma unroll
    for (int nt = 0; nt < 4; ++nt) {
      // B fragment (32x16): lane -> col = n16, 16 contiguous K at half*16
      Frag16 bf;
      bf.q[0] = *(const uint4*)&Bs[cur][(nt * 16 + n16) * 32 + half * 16];
      bf.q[1] = *(const uint4*)&Bs[cur][(nt * 16 + n16) * 32 + half * 16 + 8];
      acc[nt] = __builtin_amdgcn_wmma_f32_16x16x32_bf16(
          false, af.v, false, bf.v, (short)0, acc[nt], false, false);
    }
    __syncthreads();
#if USE_TDM
    cur ^= 1;
#endif
  }

  // C/D layout: lane -> N = n16, VGPR r -> M = r + 8*half
#pragma unroll
  for (int nt = 0; nt < 4; ++nt) {
#pragma unroll
    for (int r = 0; r < 8; ++r) {
      size_t idx = (size_t)(m0 + wave * 16 + r + 8 * half) * N + n0 + nt * 16 + n16;
      if (Cf) Cf[idx] = acc[nt][r];
      else    Cb[idx] = f2bf(acc[nt][r]);
    }
  }
}

// ---------------------------------------------------------------------------
// Fused RMSNorm + NTK-RoPE (+ optional per-head gain), in-place on bf16.
// One wave per (token, head) row of 128; T=2048 > 1024 -> NTK base rescale.
// ---------------------------------------------------------------------------
__global__ __launch_bounds__(256) void qk_norm_rope(unsigned short* __restrict__ X,
                                                    const float* __restrict__ gain,
                                                    int nheads, int nrows) {
  int wave = threadIdx.x >> 5, lane = threadIdx.x & 31;
  int row = blockIdx.x * 8 + wave;
  if (row >= nrows) return;
  size_t base = (size_t)row * 128;
  int h = row % nheads;
  int t = (row / nheads) & 2047;     // token index within T=2048

  float x1a = bf2f(X[base + lane]);
  float x1b = bf2f(X[base + lane + 32]);
  float x2a = bf2f(X[base + lane + 64]);
  float x2b = bf2f(X[base + lane + 96]);

  float ss = x1a * x1a + x1b * x1b + x2a * x2a + x2b * x2b;
#pragma unroll
  for (int m = 16; m >= 1; m >>= 1) ss += __shfl_xor(ss, m, 32);

  float r = rsqrtf(ss * (1.0f / 128.0f) + 1.1920929e-07f);   // FLT_EPSILON
  if (gain) r *= gain[h];

  // NTK: base = 1e4 * (T/train)^(D/(D-2)) = 1e4 * 2^(128/126)
  const float ntk_base = 10000.0f * __powf(2.0f, 128.0f / 126.0f);
  float fa = (float)t * __powf(ntk_base, -((float)lane) * (1.0f / 64.0f));
  float fb = (float)t * __powf(ntk_base, -((float)(lane + 32)) * (1.0f / 64.0f));
  float sa, ca, sb, cb;
  __sincosf(fa, &sa, &ca);
  __sincosf(fb, &sb, &cb);

  X[base + lane]      = f2bf(( x1a * ca + x2a * sa) * r);
  X[base + lane + 64] = f2bf((-x1a * sa + x2a * ca) * r);
  X[base + lane + 32] = f2bf(( x1b * cb + x2b * sb) * r);
  X[base + lane + 96] = f2bf((-x1b * sb + x2b * cb) * r);
}

// ---------------------------------------------------------------------------
// V post: copy f32 V into d_out's second segment and produce bf16 V for SDPA.
// ---------------------------------------------------------------------------
__global__ __launch_bounds__(256) void v_post(const float* __restrict__ Vf,
                                              float* __restrict__ vout,
                                              unsigned short* __restrict__ Vb,
                                              int n4) {
  int i = blockIdx.x * 256 + threadIdx.x;
  if (i >= n4) return;
  float4 f = ((const float4*)Vf)[i];
  ((float4*)vout)[i] = f;
  unsigned short o[4] = { f2bf(f.x), f2bf(f.y), f2bf(f.z), f2bf(f.w) };
  ((uint2*)Vb)[i] = *(const uint2*)o;
}

// ---------------------------------------------------------------------------
// Flash attention (causal, GQA 4:1). Block = 128 threads = 4 waves, covering a
// 64-row Q strip of one (b, head). KV tiles of 32 keys staged in LDS (V
// transposed for contiguous PV B-fragments). Online softmax in f32 via a
// per-wave LDS score tile (handles the C-layout -> A-layout transpose).
// ---------------------------------------------------------------------------
__global__ __launch_bounds__(128) void flash_attn(
    const unsigned short* __restrict__ Qb,   // [B*T, 2048] bf16
    const unsigned short* __restrict__ Kb,   // [B*T, 512]  bf16
    const unsigned short* __restrict__ Vb,   // [B*T, 512]  bf16
    unsigned short* __restrict__ Yb)         // [B*T, 2048] bf16
{
  constexpr int T = 2048;
  __shared__ __align__(16) unsigned short Ks[32 * 128];   // [key][d]
  __shared__ __align__(16) unsigned short Vt[128 * 32];   // [d][key]
  __shared__ __align__(16) float          Ssm[4][16 * 32];
  __shared__ __align__(16) unsigned short Psm[4][16 * 32];
  __shared__ float mrow[4][16], lrow[4][16], arow[4][16];

  int tid = threadIdx.x, wave = tid >> 5, lane = tid & 31;
  int n16 = lane & 15, half = lane >> 4;
  int q0 = blockIdx.x * 64, h = blockIdx.y, b = blockIdx.z;
  int kvh = h >> 2;
  int qw = q0 + wave * 16;

  // Q fragments (A layout), K-dim = 128 in 4 chunks of 32
  Frag16 qf[4];
  {
    size_t qrow = (size_t)(b * T + qw + n16) * 2048 + (size_t)h * 128;
#pragma unroll
    for (int kc = 0; kc < 4; ++kc) {
      qf[kc].q[0] = *(const uint4*)&Qb[qrow + kc * 32 + half * 16];
      qf[kc].q[1] = *(const uint4*)&Qb[qrow + kc * 32 + half * 16 + 8];
    }
  }

  v8f acc[8] = {};
  if (lane < 16) { mrow[wave][lane] = -3.0e38f; lrow[wave][lane] = 0.0f; }

  const float scale = 0.08838834764831845f;  // 1/sqrt(128)
  const int kend = q0 + 63;

  for (int k0 = 0; k0 <= kend; k0 += 32) {
    // Cooperative stage: K tile row-major, V tile transposed
#pragma unroll
    for (int i = 0; i < 8; ++i) {
      int e = tid * 4 + i * 512;               // 0..4095
      int key = e >> 7, d = e & 127;
      size_t g = (size_t)(b * T + k0 + key) * 512 + (size_t)kvh * 128 + d;
      *(uint2*)&Ks[key * 128 + d] = *(const uint2*)&Kb[g];
      uint2 vv = *(const uint2*)&Vb[g];
      const unsigned short* p = (const unsigned short*)&vv;
      Vt[(d + 0) * 32 + key] = p[0];
      Vt[(d + 1) * 32 + key] = p[1];
      Vt[(d + 2) * 32 + key] = p[2];
      Vt[(d + 3) * 32 + key] = p[3];
    }
    __syncthreads();

    if (k0 <= qw + 15) {
      // S = Q * K^T : two 16-key column tiles, 4 K-dim chunks each
      v8f s0 = {}, s1 = {};
#pragma unroll
      for (int kc = 0; kc < 4; ++kc) {
        Frag16 b0, b1;
        b0.q[0] = *(const uint4*)&Ks[(n16)*128 + kc * 32 + half * 16];
        b0.q[1] = *(const uint4*)&Ks[(n16)*128 + kc * 32 + half * 16 + 8];
        b1.q[0] = *(const uint4*)&Ks[(16 + n16) * 128 + kc * 32 + half * 16];
        b1.q[1] = *(const uint4*)&Ks[(16 + n16) * 128 + kc * 32 + half * 16 + 8];
        s0 = __builtin_amdgcn_wmma_f32_16x16x32_bf16(false, qf[kc].v, false, b0.v,
                                                     (short)0, s0, false, false);
        s1 = __builtin_amdgcn_wmma_f32_16x16x32_bf16(false, qf[kc].v, false, b1.v,
                                                     (short)0, s1, false, false);
      }
      // Scale + causal mask, spill to per-wave LDS score tile
#pragma unroll
      for (int r = 0; r < 8; ++r) {
        int grow = qw + r + 8 * half;
        Ssm[wave][(r + 8 * half) * 32 + n16] =
            (k0 + n16 <= grow) ? s0[r] * scale : -1.0e30f;
        Ssm[wave][(r + 8 * half) * 32 + 16 + n16] =
            (k0 + 16 + n16 <= grow) ? s1[r] * scale : -1.0e30f;
      }
      // Online softmax: lanes 0..15 each own one query row
      if (lane < 16) {
        int row = lane;
        float mo = mrow[wave][row], lo = lrow[wave][row];
        float mt = -3.0e38f;
#pragma unroll
        for (int j = 0; j < 32; ++j) mt = fmaxf(mt, Ssm[wave][row * 32 + j]);
        float mn = fmaxf(mo, mt);
        float alpha = __expf(mo - mn);
        float sum = 0.0f;
#pragma unroll
        for (int j = 0; j < 32; ++j) {
          float s = Ssm[wave][row * 32 + j];
          float pv = (s > -5.0e29f) ? __expf(s - mn) : 0.0f;
          sum += pv;
          Psm[wave][row * 32 + j] = f2bf(pv);
        }
        mrow[wave][row] = mn;
        lrow[wave][row] = alpha * lo + sum;
        arow[wave][row] = alpha;
      }
      // Rescale accumulators by per-row alpha
#pragma unroll
      for (int r = 0; r < 8; ++r) {
        float a = arow[wave][r + 8 * half];
#pragma unroll
        for (int c = 0; c < 8; ++c) acc[c][r] *= a;
      }
      // y += P * V : P as A fragment (16x32), V^T tiles as B fragments
      Frag16 pf;
      pf.q[0] = *(const uint4*)&Psm[wave][n16 * 32 + half * 16];
      pf.q[1] = *(const uint4*)&Psm[wave][n16 * 32 + half * 16 + 8];
#pragma unroll
      for (int c = 0; c < 8; ++c) {
        Frag16 vf;
        int d = c * 16 + n16;
        vf.q[0] = *(const uint4*)&Vt[d * 32 + half * 16];
        vf.q[1] = *(const uint4*)&Vt[d * 32 + half * 16 + 8];
        acc[c] = __builtin_amdgcn_wmma_f32_16x16x32_bf16(false, pf.v, false, vf.v,
                                                         (short)0, acc[c], false, false);
      }
    }
    __syncthreads();
  }

  // Finalize: divide by l, store bf16 Y
#pragma unroll
  for (int r = 0; r < 8; ++r) {
    float inv = 1.0f / lrow[wave][r + 8 * half];
    size_t orow = (size_t)(b * T + qw + r + 8 * half) * 2048 + (size_t)h * 128;
#pragma unroll
    for (int c = 0; c < 8; ++c)
      Yb[orow + c * 16 + n16] = f2bf(acc[c][r] * inv);
  }
}

// ---------------------------------------------------------------------------
// Host-side orchestration
// ---------------------------------------------------------------------------
extern "C" void kernel_launch(void* const* d_in, const int* in_sizes, int n_in,
                              void* d_out, int out_size, void* d_ws, size_t ws_size,
                              hipStream_t stream) {
  (void)in_sizes; (void)n_in; (void)out_size; (void)ws_size;
  constexpr int B = 2, T = 2048, DIM = 2048, H = 16, HKV = 4;
  constexpr int BT = B * T;
  constexpr int KVD = HKV * 128;   // 512

  const float* x  = (const float*)d_in[0];
  const float* Wq = (const float*)d_in[1];
  const float* Wk = (const float*)d_in[2];
  const float* Wv = (const float*)d_in[3];
  const float* Wp = (const float*)d_in[4];
  const float* qg = (const float*)d_in[5];

  char* w = (char*)d_ws;
  auto take = [&](size_t bytes) {
    char* p = w;
    w += (bytes + 255) & ~(size_t)255;
    return p;
  };
  unsigned short* xb  = (unsigned short*)take((size_t)BT * DIM * 2);
  unsigned short* Wqb = (unsigned short*)take((size_t)DIM * DIM * 2);
  unsigned short* Wkb = (unsigned short*)take((size_t)KVD * DIM * 2);
  unsigned short* Wvb = (unsigned short*)take((size_t)KVD * DIM * 2);
  unsigned short* Wpb = (unsigned short*)take((size_t)DIM * DIM * 2);
  unsigned short* Qb  = (unsigned short*)take((size_t)BT * DIM * 2);
  unsigned short* Kb  = (unsigned short*)take((size_t)BT * KVD * 2);
  float*          Vf  = (float*)take((size_t)BT * KVD * 4);
  unsigned short* Vb  = (unsigned short*)take((size_t)BT * KVD * 2);
  unsigned short* Yb  = (unsigned short*)take((size_t)BT * DIM * 2);

  auto cvt = [&](const float* s, unsigned short* d, size_t n) {
    int n4 = (int)(n / 4);
    cvt_f32_bf16<<<dim3((n4 + 255) / 256), dim3(256), 0, stream>>>(s, d, n4);
  };
  cvt(x,  xb,  (size_t)BT * DIM);
  cvt(Wq, Wqb, (size_t)DIM * DIM);
  cvt(Wk, Wkb, (size_t)KVD * DIM);
  cvt(Wv, Wvb, (size_t)KVD * DIM);
  cvt(Wp, Wpb, (size_t)DIM * DIM);

  // QKV projections
  gemm_bf16_nt<<<dim3(DIM / 64, BT / 128), dim3(256), 0, stream>>>(
      xb, Wqb, nullptr, Qb, BT, DIM, DIM);
  gemm_bf16_nt<<<dim3(KVD / 64, BT / 128), dim3(256), 0, stream>>>(
      xb, Wkb, nullptr, Kb, BT, KVD, DIM);
  gemm_bf16_nt<<<dim3(KVD / 64, BT / 128), dim3(256), 0, stream>>>(
      xb, Wvb, Vf, nullptr, BT, KVD, DIM);

  // QK RMSNorm + RoPE (+ q gain)
  qk_norm_rope<<<dim3(BT * H / 8), dim3(256), 0, stream>>>(Qb, qg, H, BT * H);
  qk_norm_rope<<<dim3(BT * HKV / 8), dim3(256), 0, stream>>>(Kb, nullptr, HKV, BT * HKV);

  // V -> d_out second segment (f32) and bf16 for attention
  float* vout = (float*)d_out + (size_t)BT * DIM;
  v_post<<<dim3(BT * KVD / 4 / 256), dim3(256), 0, stream>>>(Vf, vout, Vb, BT * KVD / 4);

  // Causal GQA flash attention
  flash_attn<<<dim3(T / 64, H, B), dim3(128), 0, stream>>>(Qb, Kb, Vb, Yb);

  // Output projection -> d_out first segment (f32)
  gemm_bf16_nt<<<dim3(DIM / 64, BT / 128), dim3(256), 0, stream>>>(
      Yb, Wpb, (float*)d_out, nullptr, BT, DIM, DIM);
}
